// NeuralIntraAttention_50697793962857
// MI455X (gfx1250) — matrix operand order repaired and checked by reading, not compiled, not run on hardware
//
#include <hip/hip_runtime.h>
#include <hip/hip_bf16.h>

#define VOCABN   50257
#define EXTRAN   64
#define VPN      (VOCABN + EXTRAN)   // 50321
#define SEQN     1024
#define TDECN    100
#define EDIMN    128
#define HNN      160                  // encoder hidden per direction
#define H2N      320                  // decoder hidden / 2H
#define H6N      960                  // 6H
#define G1N      640                  // 4H  (encoder gates)
#define G2N      1280                 // 8H  (decoder gates)
#define UNKTOK   3
#define STARTTOK 0
#define NEGF     (-1e9f)

typedef __attribute__((ext_vector_type(16))) __bf16          v16bf;
typedef __attribute__((ext_vector_type(16))) unsigned short  v16us;
typedef __attribute__((ext_vector_type(8)))  float           v8f;

// ---------- small device helpers ----------
__device__ __forceinline__ unsigned short f2bf(float f) {
  unsigned u = __float_as_uint(f);
  unsigned r = u + 0x7fffu + ((u >> 16) & 1u);      // round-to-nearest-even
  return (unsigned short)(r >> 16);
}
__device__ __forceinline__ float bf2f(unsigned hs) {
  return __uint_as_float((hs & 0xffffu) << 16);
}
__device__ __forceinline__ unsigned ordf(float f) { // order-preserving float->uint
  unsigned u = __float_as_uint(f);
  return (u & 0x80000000u) ? ~u : (u | 0x80000000u);
}
__device__ __forceinline__ float unordf(unsigned u) {
  return (u & 0x80000000u) ? __uint_as_float(u & 0x7fffffffu) : __uint_as_float(~u);
}
__device__ __forceinline__ float sigf(float x) { return 1.f / (1.f + __expf(-x)); }

// block reductions (blockDim.x must be a power of two)
__device__ __forceinline__ float bred_max(float v, float* red) {
  int tid = threadIdx.x;
  red[tid] = v; __syncthreads();
  for (int s = blockDim.x >> 1; s > 0; s >>= 1) {
    if (tid < s) red[tid] = fmaxf(red[tid], red[tid + s]);
    __syncthreads();
  }
  float r = red[0]; __syncthreads(); return r;
}
__device__ __forceinline__ float bred_sum(float v, float* red) {
  int tid = threadIdx.x;
  red[tid] = v; __syncthreads();
  for (int s = blockDim.x >> 1; s > 0; s >>= 1) {
    if (tid < s) red[tid] += red[tid + s];
    __syncthreads();
  }
  float r = red[0]; __syncthreads(); return r;
}

// ---------- prep kernels ----------
__global__ void k_f2bf(const float* __restrict__ in, unsigned short* __restrict__ out, int n) {
  int i = blockIdx.x * 256 + threadIdx.x;
  if (i < n) out[i] = f2bf(in[i]);
}
// transpose [R,C] f32 -> [C,R] bf16
__global__ void k_transpose_bf(const float* __restrict__ in, unsigned short* __restrict__ out,
                               int R, int C) {
  int i = blockIdx.x * 256 + threadIdx.x;
  if (i >= R * C) return;
  int r = i / C, c = i % C;
  out[(size_t)c * R + r] = f2bf(in[i]);
}
__global__ void k_vadd(const float* __restrict__ a, const float* __restrict__ b,
                       float* __restrict__ o, int n) {
  int i = blockIdx.x * 256 + threadIdx.x;
  if (i < n) o[i] = a[i] + b[i];
}
__global__ void k_zero(float* __restrict__ p, int n) {
  int i = blockIdx.x * 256 + threadIdx.x;
  if (i < n) p[i] = 0.f;
}
__global__ void k_gather_emb(const int* __restrict__ ids, const float* __restrict__ embedding,
                             unsigned short* __restrict__ eb) {
  int i = blockIdx.x * 256 + threadIdx.x;
  if (i >= SEQN * EDIMN) return;
  int s = i >> 7, e = i & 127;
  int id = ids[s]; if (id >= VOCABN) id = UNKTOK;
  eb[i] = f2bf(embedding[(size_t)id * EDIMN + e]);
}

// ---------- WMMA bf16 GEMM: C[M,N] = op(A[M,K] * Bt[N,K]^T (+bias)) ----------
// B is supplied N-MAJOR (Bt[n][k]) so both A and B stage with b128 loads/stores.
// Exploited: N % 64 == 0, K % 32 == 0, lda/ldbt % 8 == 0, 16B-aligned bases.
// Only M is ragged (row-clamped loads, guarded stores).
// Block = 256 threads (8 waves), block tile 128(M) x 64(N); each wave owns a
// 32x32 output = 4 WMMA tiles from 2 A + 2 B fragments (1.0 LDS frag loads/WMMA).
// LDS is pre-arranged in WMMA fragment order (ISA 7.12.2, wave32):
//   A fragment element i of half-wave kh reads K = ((i<8)?i:i+8) + kh*8
//     -> As_p[m][slot], k-group permutation 0,2,1,3: each lane's 16 elements are
//        32 contiguous bytes (two ds_load_b128).
//   B fragment element i reads the same K as A, at fixed N per lane
//     -> Bs_t[n][slot] (N-major), same k-group permutation 0,2,1,3 as A.
#define TM 128
#define TN 64
#define TK 32
#define LDST 48   // LDS row stride (ushorts); 96B keeps 32B fragment alignment
__global__ __launch_bounds__(256) void k_gemm_bf16(
    const unsigned short* __restrict__ A, int lda,
    const unsigned short* __restrict__ Bt, int ldbt,
    const float* __restrict__ bias,
    float* __restrict__ Cf, unsigned short* __restrict__ Cb,
    int M, int N, int K, int ldc, int do_tanh)
{
  __shared__ unsigned short As_p[TM][LDST];
  __shared__ unsigned short Bs_t[TN][LDST];
  int t = threadIdx.x;
  int lane = t & 31, wave = t >> 5;
  int mbase = blockIdx.y * TM;
  int nbase = blockIdx.x * TN;
  int wm = (wave >> 1) * 32;          // 0,32,64,96
  int wn = (wave & 1) * 32;           // 0,32
  v8f acc00 = {}, acc01 = {}, acc10 = {}, acc11 = {};

  // staging indices (uniform, no divergence)
  int ar  = t >> 1;                   // A row 0..127 (2 threads per row)
  int ak  = (t & 1) * 16;             // A k-offset: 0 or 16 (two groups of 8)
  int as0 = (t & 1) ? 8 : 0;          // permuted slot for first group (g0->0, g2->8)
  int bn  = t >> 2;                   // B n-row 0..63
  int bk  = (t & 3) * 8;              // B k-offset (multiple of 8)

  int khalf = lane >> 4;
  int l15   = lane & 15;

  #pragma unroll 2
  for (int kb = 0; kb < K; kb += TK) {
    __syncthreads();
    // stage A tile 128x32: two uint4 per thread, permuted into fragment order
    {
      int gm = mbase + ar; if (gm >= M) gm = M - 1;
      const unsigned short* ap = A + (size_t)gm * lda + kb + ak;
      uint4 va0 = *(const uint4*)(ap);
      uint4 va1 = *(const uint4*)(ap + 8);
      *(uint4*)&As_p[ar][as0]      = va0;   // groups: 0->slot0, 2->slot8
      *(uint4*)&As_p[ar][as0 + 16] = va1;   //         1->slot16, 3->slot24
    }
    // stage B tile 64x32: one uint4 per thread, already N-major
    {
      uint4 vb = *(const uint4*)(Bt + (size_t)(nbase + bn) * ldbt + kb + bk);
      *(uint4*)&Bs_t[bn][((t & 1) << 4) | ((t & 2) << 2)] = vb;
    }
    __syncthreads();

    // fragment loads: 32 contiguous bytes each (2x ds_load_b128)
    v16us au0 = *(const v16us*)&As_p[wm + l15][khalf * 16];
    v16us au1 = *(const v16us*)&As_p[wm + 16 + l15][khalf * 16];
    v16us bu0 = *(const v16us*)&Bs_t[wn + l15][khalf * 16];
    v16us bu1 = *(const v16us*)&Bs_t[wn + 16 + l15][khalf * 16];
    v16bf a0 = __builtin_bit_cast(v16bf, au0);
    v16bf a1 = __builtin_bit_cast(v16bf, au1);
    v16bf b0 = __builtin_bit_cast(v16bf, bu0);
    v16bf b1 = __builtin_bit_cast(v16bf, bu1);
    acc00 = __builtin_amdgcn_wmma_f32_16x16x32_bf16(false, a0, false, b0,
                                                    (short)0, acc00, false, false);
    acc01 = __builtin_amdgcn_wmma_f32_16x16x32_bf16(false, a0, false, b1,
                                                    (short)0, acc01, false, false);
    acc10 = __builtin_amdgcn_wmma_f32_16x16x32_bf16(false, a1, false, b0,
                                                    (short)0, acc10, false, false);
    acc11 = __builtin_amdgcn_wmma_f32_16x16x32_bf16(false, a1, false, b1,
                                                    (short)0, acc11, false, false);
  }

  // epilogue: C/D layout: m = i + (lane>>4)*8, n = lane&15
  int n0 = nbase + wn + l15;
  int n1 = n0 + 16;
  float bv0 = bias ? bias[n0] : 0.f;
  float bv1 = bias ? bias[n1] : 0.f;
  #pragma unroll
  for (int i = 0; i < 8; ++i) {
    int m0 = mbase + wm + i + khalf * 8;
    int m1 = m0 + 16;
    float v00 = acc00[i] + bv0, v01 = acc01[i] + bv1;
    float v10 = acc10[i] + bv0, v11 = acc11[i] + bv1;
    if (do_tanh) {
      v00 = tanhf(v00); v01 = tanhf(v01); v10 = tanhf(v10); v11 = tanhf(v11);
    }
    if (m0 < M) {
      if (Cf) { Cf[(size_t)m0 * ldc + n0] = v00; Cf[(size_t)m0 * ldc + n1] = v01; }
      if (Cb) { Cb[(size_t)m0 * ldc + n0] = f2bf(v00); Cb[(size_t)m0 * ldc + n1] = f2bf(v01); }
    }
    if (m1 < M) {
      if (Cf) { Cf[(size_t)m1 * ldc + n0] = v10; Cf[(size_t)m1 * ldc + n1] = v11; }
      if (Cb) { Cb[(size_t)m1 * ldc + n0] = f2bf(v10); Cb[(size_t)m1 * ldc + n1] = f2bf(v11); }
    }
  }
}

// ---------- encoder: sequential bidirectional LSTM (1 block per direction) ----------
__global__ __launch_bounds__(640) void k_encoder(
    const float* __restrict__ xw_f, const float* __restrict__ xw_b,
    const float* __restrict__ whh_f, const float* __restrict__ whh_b,
    float* __restrict__ enc_h, float* __restrict__ dec_h)
{
  __shared__ float hs[HNN];
  __shared__ float cs[HNN];
  __shared__ float gs[G1N];
  int t = threadIdx.x;                 // 640
  int dir = blockIdx.x;                // 0 = fwd, 1 = bwd
  const float* xw  = dir ? xw_b  : xw_f;
  const float* whh = dir ? whh_b : whh_f;
  if (t < HNN) { hs[t] = 0.f; cs[t] = 0.f; }
  __syncthreads();
  for (int step = 0; step < SEQN; ++step) {
    int s = dir ? (SEQN - 1 - step) : step;
    float acc = xw[(size_t)s * G1N + t];           // x@Wih^T + bih + bhh precomputed
    const float* wr = whh + (size_t)t * HNN;
    #pragma unroll 4
    for (int k = 0; k < HNN; ++k) acc += wr[k] * hs[k];
    gs[t] = acc;
    __syncthreads();
    if (t < HNN) {
      float ig = sigf(gs[t]);
      float fg = sigf(gs[HNN + t]);
      float gg = tanhf(gs[2 * HNN + t]);
      float og = sigf(gs[3 * HNN + t]);
      float c = fg * cs[t] + ig * gg;
      float h = og * tanhf(c);
      cs[t] = c; hs[t] = h;
      enc_h[(size_t)s * H2N + dir * HNN + t] = h;
    }
    __syncthreads();
  }
  if (t < HNN) dec_h[dir * HNN + t] = hs[t];       // h0 = [hfin_f, hfin_b]
}

// ---------- decoder step (sequential part): 1 block, 512 threads ----------
__global__ __launch_bounds__(512) void k_dec_step(
    int t,
    const float* __restrict__ embedding,
    const float* __restrict__ dec_wih, const float* __restrict__ dec_whh,
    const float* __restrict__ dec_bih, const float* __restrict__ dec_bhh,
    const float* __restrict__ enc_proj, const float* __restrict__ enc_h,
    const float* __restrict__ dap,
    const float* __restrict__ switch_w, const float* __restrict__ switch_b,
    float* __restrict__ dec_h, float* __restrict__ dec_c,
    float* __restrict__ cum, float* __restrict__ dec_buf,
    float* __restrict__ attn_g, unsigned short* __restrict__ concat_bf,
    float* __restrict__ p_copy_g,
    unsigned* __restrict__ gmax, float* __restrict__ gsum,
    unsigned long long* __restrict__ argmax_g)
{
  __shared__ float x_s[EDIMN];
  __shared__ float hprev[H2N];
  __shared__ float hnew[H2N];
  __shared__ float gates[G2N];
  __shared__ float sc_s[SEQN];
  __shared__ float dq_s[H2N];
  __shared__ float conc[H6N];
  __shared__ float dsc[128];
  __shared__ float red[512];
  int tid = threadIdx.x;

  // greedy token from previous step's argmax
  int tok;
  if (t == 0) tok = STARTTOK;
  else {
    unsigned long long key = *argmax_g;
    tok = (int)(0xffffffffu - (unsigned)(key & 0xffffffffull));
  }
  if (tok >= VOCABN || tok < 0) tok = UNKTOK;

  if (tid < EDIMN) x_s[tid] = embedding[(size_t)tok * EDIMN + tid];
  if (tid < H2N)   hprev[tid] = dec_h[tid];
  __syncthreads();

  // LSTM gates (1280 outputs, dot length 448)
  for (int j = tid; j < G2N; j += 512) {
    float acc = dec_bih[j] + dec_bhh[j];
    const float* wi = dec_wih + (size_t)j * EDIMN;
    #pragma unroll 4
    for (int k = 0; k < EDIMN; ++k) acc += wi[k] * x_s[k];
    const float* wh = dec_whh + (size_t)j * H2N;
    #pragma unroll 4
    for (int k = 0; k < H2N; ++k) acc += wh[k] * hprev[k];
    gates[j] = acc;
  }
  __syncthreads();

  if (tid < H2N) {
    float ig = sigf(gates[tid]);
    float fg = sigf(gates[H2N + tid]);
    float gg = tanhf(gates[2 * H2N + tid]);
    float og = sigf(gates[3 * H2N + tid]);
    float c = fg * dec_c[tid] + ig * gg;
    float h = og * tanhf(c);
    dec_c[tid] = c;
    hnew[tid]  = h;
    dec_h[tid] = h;
    dec_buf[(size_t)t * H2N + tid] = h;
    conc[tid] = h;
  }
  __syncthreads();

  // intra-temporal encoder attention: scores, temporal rescale, cum update
  for (int s = tid; s < SEQN; s += 512) {
    const float* ep = enc_proj + (size_t)s * H2N;
    float acc = 0.f;
    #pragma unroll 4
    for (int k = 0; k < H2N; ++k) acc += ep[k] * hnew[k];
    float tempo = (t == 0) ? acc : (__expf(acc) / cum[s]);
    cum[s] += acc;
    sc_s[s] = tempo;
  }
  __syncthreads();

  // softmax over 1024 temporal scores
  float lm = -3.4e38f;
  for (int s = tid; s < SEQN; s += 512) lm = fmaxf(lm, sc_s[s]);
  lm = bred_max(lm, red);
  float ls = 0.f;
  for (int s = tid; s < SEQN; s += 512) { float e = __expf(sc_s[s] - lm); sc_s[s] = e; ls += e; }
  ls = bred_sum(ls, red);
  float linv = 1.f / ls;
  for (int s = tid; s < SEQN; s += 512) { float a = sc_s[s] * linv; sc_s[s] = a; attn_g[s] = a; }
  __syncthreads();

  // enc_ctx and decoder-attention query
  if (tid < H2N) {
    float acc = 0.f;
    for (int s = 0; s < SEQN; ++s) acc += sc_s[s] * enc_h[(size_t)s * H2N + tid];
    conc[H2N + tid] = acc;
    float dacc = 0.f;
    for (int k = 0; k < H2N; ++k) dacc += hnew[k] * dap[(size_t)k * H2N + tid];
    dq_s[tid] = dacc;
  }
  __syncthreads();

  // intra-decoder attention over previous hidden states
  if (tid < 128) {
    float v = NEGF;
    if (tid < t) {
      const float* db = dec_buf + (size_t)tid * H2N;
      float acc = 0.f;
      for (int k = 0; k < H2N; ++k) acc += dq_s[k] * db[k];
      v = acc;
    }
    dsc[tid] = v;
  }
  __syncthreads();
  float dm = bred_max((tid < 128) ? dsc[tid] : NEGF, red);
  float de = (tid < 128) ? __expf(dsc[tid] - dm) : 0.f;
  float dZ = bred_sum(de, red);
  if (tid < 128) dsc[tid] = de / dZ;
  __syncthreads();
  if (tid < H2N) {
    float acc = 0.f;
    for (int u = 0; u < t; ++u) acc += dsc[u] * dec_buf[(size_t)u * H2N + tid];
    conc[2 * H2N + tid] = (t == 0) ? 0.f : acc;
  }
  __syncthreads();

  // switch probability + export concat (bf16 for the L2-resident vocab matvec)
  float sp = 0.f;
  for (int i = tid; i < H6N; i += 512) sp += switch_w[i] * conc[i];
  sp = bred_sum(sp, red);
  for (int i = tid; i < H6N; i += 512) concat_bf[i] = f2bf(conc[i]);
  if (tid == 0) {
    *p_copy_g = sigf(sp + switch_b[0]);
    *gmax = 0u;          // ordered-uint lower bound
    *gsum = 0.f;
    *argmax_g = 0ull;
  }
}

// ---------- vocab logits: wave-per-row bf16 GEMV (out_proj lives in L2) ----------
__global__ __launch_bounds__(256) void k_logits(
    const unsigned* __restrict__ out_proj,       // bf16 pairs, [V][480] u32
    const unsigned* __restrict__ concat_u,       // bf16 pairs, [480] u32
    const float* __restrict__ out_bias,
    float* __restrict__ z, unsigned* __restrict__ gmax)
{
  __shared__ float cc[H6N];
  __shared__ float wmax[8];
  int tid = threadIdx.x;
  int lane = tid & 31, w = tid >> 5;
  for (int i = tid; i < H6N / 2; i += 256) {
    unsigned u = concat_u[i];
    cc[2 * i]     = bf2f(u & 0xffffu);
    cc[2 * i + 1] = bf2f(u >> 16);
  }
  __syncthreads();
  int v = blockIdx.x * 8 + w;
  float acc = 0.f;
  if (v < VOCABN) {
    const unsigned* row = out_proj + (size_t)v * (H6N / 2);
    __builtin_prefetch(row + 240, 0, 0);         // global_prefetch: back half of row
    for (int i = lane; i < H6N / 2; i += 32) {
      unsigned u = row[i];
      acc += bf2f(u & 0xffffu) * cc[2 * i] + bf2f(u >> 16) * cc[2 * i + 1];
    }
  }
  for (int off = 16; off > 0; off >>= 1) acc += __shfl_down(acc, off, 32);
  if (lane == 0) {
    float zv = -3.4e38f;
    if (v < VOCABN) { zv = acc + out_bias[v]; z[v] = zv; }
    wmax[w] = zv;
  }
  __syncthreads();
  if (tid == 0) {
    float m = wmax[0];
    #pragma unroll
    for (int k = 1; k < 8; ++k) m = fmaxf(m, wmax[k]);
    atomicMax(gmax, ordf(m));
  }
}

__global__ void k_expsum(const float* __restrict__ z, const unsigned* __restrict__ gmax,
                         float* __restrict__ outrow, float* __restrict__ gsum)
{
  __shared__ float red[256];
  int i = blockIdx.x * 256 + threadIdx.x;
  float m = unordf(*gmax);
  float e = 0.f;
  if (i < VOCABN)   { e = __expf(z[i] - m); outrow[i] = e; }
  else if (i < VPN) { outrow[i] = 0.f; }
  float s = bred_sum(e, red);
  if (threadIdx.x == 0) atomicAdd(gsum, s);
}

__global__ void k_norm(float* __restrict__ outrow, const float* __restrict__ p_copy,
                       const float* __restrict__ gsum)
{
  int i = blockIdx.x * 256 + threadIdx.x;
  if (i < VOCABN) outrow[i] *= (1.f - *p_copy) / *gsum;
}

__global__ void k_scatter(float* __restrict__ outrow, const int* __restrict__ ids,
                          const float* __restrict__ attn, const float* __restrict__ p_copy)
{
  int s = threadIdx.x;   // 1024
  atomicAdd(&outrow[ids[s]], (*p_copy) * attn[s]);
}

__global__ void k_argmax(const float* __restrict__ outrow,
                         unsigned long long* __restrict__ argmax_g)
{
  __shared__ unsigned long long r64[256];
  int i = blockIdx.x * 256 + threadIdx.x;
  unsigned long long key = 0ull;
  if (i < VPN)
    key = ((unsigned long long)ordf(outrow[i]) << 32) |
          (unsigned long long)(0xffffffffu - (unsigned)i);   // first-max tie-break
  r64[threadIdx.x] = key; __syncthreads();
  for (int s = 128; s > 0; s >>= 1) {
    if (threadIdx.x < s && r64[threadIdx.x + s] > r64[threadIdx.x])
      r64[threadIdx.x] = r64[threadIdx.x + s];
    __syncthreads();
  }
  if (threadIdx.x == 0) atomicMax(argmax_g, r64[0]);
}

// ---------- host ----------
extern "C" void kernel_launch(void* const* d_in, const int* in_sizes, int n_in,
                              void* d_out, int out_size, void* d_ws, size_t ws_size,
                              hipStream_t stream)
{
  (void)in_sizes; (void)n_in; (void)out_size; (void)ws_size;
  const int*   ids       = (const int*)  d_in[0];
  const float* embedding = (const float*)d_in[1];
  const float* wih_f     = (const float*)d_in[2];
  const float* whh_f     = (const float*)d_in[3];
  const float* bih_f     = (const float*)d_in[4];
  const float* bhh_f     = (const float*)d_in[5];
  const float* wih_b     = (const float*)d_in[6];
  const float* whh_b     = (const float*)d_in[7];
  const float* bih_b     = (const float*)d_in[8];
  const float* bhh_b     = (const float*)d_in[9];
  const float* dwih      = (const float*)d_in[10];
  const float* dwhh      = (const float*)d_in[11];
  const float* dbih      = (const float*)d_in[12];
  const float* dbhh      = (const float*)d_in[13];
  const float* eap       = (const float*)d_in[14];
  const float* dap       = (const float*)d_in[15];
  const float* vp        = (const float*)d_in[16];
  const float* out_bias  = (const float*)d_in[17];
  const float* sw        = (const float*)d_in[18];
  const float* sb        = (const float*)d_in[19];
  float* out = (float*)d_out;

  char* ws = (char*)d_ws;
  size_t o = 0;
  auto alloc = [&](size_t bytes) { size_t r = o; o += (bytes + 255) & ~(size_t)255; return r; };

  unsigned short* embA_bf   = (unsigned short*)(ws + alloc((size_t)VOCABN * EDIMN * 2));
  unsigned short* vpT_bf    = (unsigned short*)(ws + alloc((size_t)H6N * EDIMN * 2));   // [960][128] N-major
  unsigned short* wih_f_bf  = (unsigned short*)(ws + alloc((size_t)G1N * EDIMN * 2));   // [640][128] N-major
  unsigned short* wih_b_bf  = (unsigned short*)(ws + alloc((size_t)G1N * EDIMN * 2));
  unsigned short* eap_bf    = (unsigned short*)(ws + alloc((size_t)H2N * H2N * 2));     // [320][320] N-major
  float*          bias_f    = (float*)(ws + alloc((size_t)G1N * 4));
  float*          bias_b    = (float*)(ws + alloc((size_t)G1N * 4));
  unsigned short* embids_bf = (unsigned short*)(ws + alloc((size_t)SEQN * EDIMN * 2));
  float*          xw_f      = (float*)(ws + alloc((size_t)SEQN * G1N * 4));
  float*          xw_b      = (float*)(ws + alloc((size_t)SEQN * G1N * 4));
  unsigned short* outproj   = (unsigned short*)(ws + alloc((size_t)VOCABN * H6N * 2));
  float*          enc_h     = (float*)(ws + alloc((size_t)SEQN * H2N * 4));
  unsigned short* enc_h_bf  = (unsigned short*)(ws + alloc((size_t)SEQN * H2N * 2));
  float*          enc_proj  = (float*)(ws + alloc((size_t)SEQN * H2N * 4));
  float*          dec_h     = (float*)(ws + alloc((size_t)H2N * 4));
  float*          dec_c     = (float*)(ws + alloc((size_t)H2N * 4));
  float*          cum       = (float*)(ws + alloc((size_t)SEQN * 4));
  float*          dec_buf   = (float*)(ws + alloc((size_t)TDECN * H2N * 4));
  float*          attn_g    = (float*)(ws + alloc((size_t)SEQN * 4));
  unsigned short* concat_bf = (unsigned short*)(ws + alloc((size_t)H6N * 2));
  float*          zbuf      = (float*)(ws + alloc((size_t)VOCABN * 4));
  char*           scal      = ws + alloc(256);
  float*              p_copy = (float*)(scal + 0);
  unsigned*           gmax   = (unsigned*)(scal + 64);
  float*              gsum   = (float*)(scal + 128);
  unsigned long long* argmax = (unsigned long long*)(scal + 192);

  // ---- precompute ----
  k_f2bf<<<(VOCABN * EDIMN + 255) / 256, 256, 0, stream>>>(embedding, embA_bf, VOCABN * EDIMN);
  // vocab_proj [128,960] -> N-major [960][128] bf16 (only transpose needed)
  k_transpose_bf<<<(EDIMN * H6N + 255) / 256, 256, 0, stream>>>(vp, vpT_bf, EDIMN, H6N);
  // enc_wih_* (640,128) and enc_attn_proj (320,320) are already N-major: convert only
  k_f2bf<<<(G1N * EDIMN + 255) / 256, 256, 0, stream>>>(wih_f, wih_f_bf, G1N * EDIMN);
  k_f2bf<<<(G1N * EDIMN + 255) / 256, 256, 0, stream>>>(wih_b, wih_b_bf, G1N * EDIMN);
  k_f2bf<<<(H2N * H2N + 255) / 256, 256, 0, stream>>>(eap, eap_bf, H2N * H2N);
  k_vadd<<<(G1N + 255) / 256, 256, 0, stream>>>(bih_f, bhh_f, bias_f, G1N);
  k_vadd<<<(G1N + 255) / 256, 256, 0, stream>>>(bih_b, bhh_b, bias_b, G1N);
  k_gather_emb<<<(SEQN * EDIMN + 255) / 256, 256, 0, stream>>>(ids, embedding, embids_bf);

  // out_proj = tanh(embedding @ vocab_proj)  -> bf16, L2-resident for the decode loop
  dim3 gOP(H6N / TN, (VOCABN + TM - 1) / TM);                 // 15 x 393
  k_gemm_bf16<<<gOP, 256, 0, stream>>>(embA_bf, EDIMN, vpT_bf, EDIMN, nullptr,
                                       nullptr, outproj, VOCABN, H6N, EDIMN, H6N, 1);
  // encoder input projections (x@Wih^T + bih + bhh)
  dim3 gXW(G1N / TN, SEQN / TM);                              // 10 x 8
  k_gemm_bf16<<<gXW, 256, 0, stream>>>(embids_bf, EDIMN, wih_f_bf, EDIMN, bias_f,
                                       xw_f, nullptr, SEQN, G1N, EDIMN, G1N, 0);
  k_gemm_bf16<<<gXW, 256, 0, stream>>>(embids_bf, EDIMN, wih_b_bf, EDIMN, bias_b,
                                       xw_b, nullptr, SEQN, G1N, EDIMN, G1N, 0);

  k_zero<<<(SEQN + 255) / 256, 256, 0, stream>>>(cum, SEQN);
  k_zero<<<(H2N + 255) / 256, 256, 0, stream>>>(dec_c, H2N);

  // sequential bidirectional encoder (also writes h0 into dec_h)
  k_encoder<<<2, 640, 0, stream>>>(xw_f, xw_b, whh_f, whh_b, enc_h, dec_h);

  // enc_proj_h = enc_h @ enc_attn_proj^T
  k_f2bf<<<(SEQN * H2N + 255) / 256, 256, 0, stream>>>(enc_h, enc_h_bf, SEQN * H2N);
  dim3 gEP(H2N / TN, SEQN / TM);                              // 5 x 8
  k_gemm_bf16<<<gEP, 256, 0, stream>>>(enc_h_bf, H2N, eap_bf, H2N, nullptr,
                                       enc_proj, nullptr, SEQN, H2N, H2N, H2N, 0);

  // ---- greedy decode loop (graph-captured kernel chain, no host sync) ----
  for (int t = 0; t < TDECN; ++t) {
    k_dec_step<<<1, 512, 0, stream>>>(t, embedding, dwih, dwhh, dbih, dbhh,
                                      enc_proj, enc_h, dap, sw, sb,
                                      dec_h, dec_c, cum, dec_buf, attn_g, concat_bf,
                                      p_copy, gmax, gsum, argmax);
    float* outrow = out + (size_t)t * VPN;
    k_logits<<<(VOCABN + 7) / 8, 256, 0, stream>>>((const unsigned*)outproj,
                                                   (const unsigned*)concat_bf,
                                                   out_bias, zbuf, gmax);
    k_expsum<<<(VPN + 255) / 256, 256, 0, stream>>>(zbuf, gmax, outrow, gsum);
    k_norm<<<(VOCABN + 255) / 256, 256, 0, stream>>>(outrow, p_copy, gsum);
    k_scatter<<<1, 1024, 0, stream>>>(outrow, ids, attn_g, p_copy);
    k_argmax<<<(VPN + 255) / 256, 256, 0, stream>>>(outrow, argmax);
  }
}
